// RelationalLatentDynamics_62912680952255
// MI455X (gfx1250) — matrix operand order repaired and checked by reading, not compiled, not run on hardware
//
#include <hip/hip_runtime.h>
#include <stdint.h>

// ---------------------------------------------------------------------------
// RelationalLatentDynamics fused kernel for gfx1250 (MI455X).
// One workgroup (256 threads = 8 wave32) per batch element. All weights are
// converted to f16 and pre-swizzled into LDS in WMMA B-fragment order; all
// activations and the recurrent z-state live in LDS. Every GEMM is a
// v_wmma_f32_16x16x32_f16. Attention softmax uses wave32 shfl_xor reductions.
// ---------------------------------------------------------------------------

typedef __attribute__((ext_vector_type(16))) _Float16 v16h;
typedef __attribute__((ext_vector_type(8)))  _Float16 v8h;
typedef __attribute__((ext_vector_type(8)))  float    v8f;

#define DEVINL __device__ __forceinline__

// Problem dims (fixed by the reference).
constexpr int Bn  = 256;
constexpr int Tn  = 8;
constexpr int On  = 32;
constexpr int En  = 16;
constexpr int TFn = 8;
constexpr int TEn = Tn * En;   // 128
constexpr int EIn = 32;        // E + I
constexpr float INV_TEMP = 2.0f;  // 1 / 0.5

// LDS strides in halves (all chosen so row pitches are 16B multiples).
constexpr int SRC_LD = 136;
constexpr int Z_LD   = 40;
constexpr int AGG_LD = 72;
constexpr int HB_LD  = 72;
constexpr int EB_LD  = 72;

constexpr size_t al64(size_t x) { return (x + 63) & ~(size_t)63; }

constexpr size_t OFF_SRC  = 0;                                   // 32x136 f16
constexpr size_t OFF_Z16  = al64(OFF_SRC + 32 * SRC_LD * 2);     // 32x40  f16
constexpr size_t OFF_AGG  = al64(OFF_Z16 + 32 * Z_LD * 2);       // 32x72  f16
constexpr size_t OFF_HB   = al64(OFF_AGG + 32 * AGG_LD * 2);     // 8w x 2 x 16x72 f16
constexpr size_t OFF_EB   = al64(OFF_HB + 8 * 2 * 16 * HB_LD * 2); // 8w x 32x72 f16
constexpr size_t OFF_ZF   = al64(OFF_EB + 8 * 32 * EB_LD * 2);   // 32x32 f32
constexpr size_t OFF_ATTN = al64(OFF_ZF + 32 * EIn * 4);         // 8w x 32 f32
constexpr size_t OFF_BIAS = al64(OFF_ATTN + 8 * 32 * 4);         // 1024 f32 arena
constexpr size_t OFF_W    = al64(OFF_BIAS + 1024 * 4);
// Swizzled f16 weight blocks (sizes in halves).
constexpr size_t OFF_WE1 = OFF_W;                 // 256x64
constexpr size_t OFF_WE2 = OFF_WE1 + 16384 * 2;   // 64x64
constexpr size_t OFF_WE3 = OFF_WE2 + 4096 * 2;    // 64x64
constexpr size_t OFF_WN1 = OFF_WE3 + 4096 * 2;    // 192x64
constexpr size_t OFF_WN2 = OFF_WN1 + 12288 * 2;   // 64x64
constexpr size_t OFF_WN3 = OFF_WN2 + 4096 * 2;    // 64x32
constexpr size_t OFF_WN4 = OFF_WN3 + 2048 * 2;    // 32x16
constexpr size_t OFF_WL1 = OFF_WN4 + 512 * 2;     // 64x64
constexpr size_t OFF_WL2 = OFF_WL1 + 4096 * 2;    // 64x64
constexpr size_t OFF_WL3 = OFF_WL2 + 4096 * 2;    // 64x64
constexpr size_t OFF_WT1 = OFF_WL3 + 4096 * 2;    // 96x64
constexpr size_t OFF_WT2 = OFF_WT1 + 6144 * 2;    // 64x64
constexpr size_t OFF_WT3 = OFF_WT2 + 4096 * 2;    // 64x32
constexpr size_t OFF_WT4 = OFF_WT3 + 2048 * 2;    // 32x32
constexpr size_t SMEM_BYTES = OFF_WT4 + 1024 * 2; // = 237056 B (< 320 KB/WGP)

// Bias arena offsets (float indices).
constexpr int BE1 = 0,  BE2 = 64, BE3 = 128, AGGW = 192, AGGB = 256,
              BN1 = 288, BN2 = 352, BN3 = 416, BN4 = 448,
              BL1 = 480, BL2 = 544, BL3 = 608, LAGGW = 672, LAGGB = 736,
              BT1 = 768, BT2 = 832, BT3 = 896, BT4 = 928;

struct Params { const float* p[33]; };

DEVINL v8f wmma16(v16h a, v16h b, v8f c) {
  return __builtin_amdgcn_wmma_f32_16x16x32_f16(
      /*neg_a=*/false, a, /*neg_b=*/false, b,
      /*c_mod=*/(short)0, c, /*reuse_a=*/false, /*reuse_b=*/false);
}

// A fragment (16x32 f16, MxK): lane's 16 halves are two contiguous 16B chunks
// at K = kh and K = 16+kh (kh = 0 for lanes 0-15, 8 for lanes 16-31).
DEVINL v16h load_a_frag(const _Float16* row_ptr, int lane) {
  int kh = (lane & 16) ? 8 : 0;
  union { v16h v; v8h h[2]; } u;
  u.h[0] = *(const v8h*)(row_ptr + kh);
  u.h[1] = *(const v8h*)(row_ptr + 16 + kh);
  return u.v;
}

// B fragment from pre-swizzled weights: each (kbi, nb) fragment is 32 lanes x
// 16 contiguous halves (lane's halves = K kh..kh+7, 16+kh..16+kh+7 at fixed column).
DEVINL v16h load_b_swz(const _Float16* S, int nbW, int kbi, int nb, int lane) {
  return *(const v16h*)(S + (((size_t)(kbi * nbW + nb) * 32 + lane) << 4));
}

// D epilogue: c(m,n) with m = j + 8*(lane>=16), n = n0 + (lane&15).
DEVINL void epi_store(v8f c, const float* bias, int n0, bool relu,
                      _Float16* dst, int ld, int lane) {
  int n = n0 + (lane & 15);
  float bv = bias[n];
  int mb = (lane & 16) ? 8 : 0;
#pragma unroll
  for (int j = 0; j < 8; ++j) {
    float v = c[j] + bv;
    if (relu) v = v > 0.0f ? v : 0.0f;
    dst[(mb + j) * ld + n] = (_Float16)v;
  }
}

// Swizzle a [K,N] f32 weight into f16 B-fragment order.
DEVINL void load_w_swz(const float* G, int K, int N, _Float16* S, int tid) {
  int total = K * N;
  int nbW = N >> 4;
  for (int i = tid; i < total; i += 256) {
    int j  = i & 15;
    int ln = (i >> 4) & 31;
    int fr = i >> 9;
    int kbi = fr / nbW;
    int nb  = fr - kbi * nbW;
    int col = nb * 16 + (ln & 15);
    int k   = kbi * 32 + ((ln & 16) ? 8 : 0) + (j & 7) + ((j & 8) ? 16 : 0);
    S[i] = (_Float16)G[k * N + col];
  }
}

DEVINL void copy_f32(const float* g, float* s, int n, int tid) {
  for (int i = tid; i < n; i += 256) s[i] = g[i];
}

// 3-layer pair MLP on one 16-row tile: K1=NKB*32 -> 64 relu -> 64 relu -> 64.
template <int NKB>
DEVINL void pair_mlp3(const _Float16* const* aptr,
                      const _Float16* W1, const float* b1,
                      const _Float16* W2, const float* b2,
                      const _Float16* W3, const float* b3,
                      _Float16* hb0, _Float16* hb1, _Float16* eout, int lane) {
  v8f zero = {0.f, 0.f, 0.f, 0.f, 0.f, 0.f, 0.f, 0.f};
  v8f acc[4] = {zero, zero, zero, zero};
#pragma unroll
  for (int kb = 0; kb < NKB; ++kb) {
    v16h a = load_a_frag(aptr[kb], lane);
#pragma unroll
    for (int nb = 0; nb < 4; ++nb)
      acc[nb] = wmma16(a, load_b_swz(W1, 4, kb, nb, lane), acc[nb]);
  }
#pragma unroll
  for (int nb = 0; nb < 4; ++nb)
    epi_store(acc[nb], b1, nb * 16, true, hb0, HB_LD, lane);

  {
    const _Float16* hrow = hb0 + (lane & 15) * HB_LD;
    v8f a2[4] = {zero, zero, zero, zero};
#pragma unroll
    for (int kb = 0; kb < 2; ++kb) {
      v16h a = load_a_frag(hrow + kb * 32, lane);
#pragma unroll
      for (int nb = 0; nb < 4; ++nb)
        a2[nb] = wmma16(a, load_b_swz(W2, 4, kb, nb, lane), a2[nb]);
    }
#pragma unroll
    for (int nb = 0; nb < 4; ++nb)
      epi_store(a2[nb], b2, nb * 16, true, hb1, HB_LD, lane);
  }
  {
    const _Float16* hrow = hb1 + (lane & 15) * HB_LD;
    v8f a3[4] = {zero, zero, zero, zero};
#pragma unroll
    for (int kb = 0; kb < 2; ++kb) {
      v16h a = load_a_frag(hrow + kb * 32, lane);
#pragma unroll
      for (int nb = 0; nb < 4; ++nb)
        a3[nb] = wmma16(a, load_b_swz(W3, 4, kb, nb, lane), a3[nb]);
    }
#pragma unroll
    for (int nb = 0; nb < 4; ++nb)
      epi_store(a3[nb], b3, nb * 16, false, eout, EB_LD, lane);  // last layer: no relu
  }
}

// Leading 3 layers of the 4-layer node MLPs: K1 -> 64 relu -> 64 relu -> 32 relu.
// Result (16x32, f16) ends in hb0 columns 0..31.
template <int NKB1>
DEVINL void trunk3(const _Float16* const* aptr,
                   const _Float16* W1, const float* b1,
                   const _Float16* W2, const float* b2,
                   const _Float16* W3, const float* b3,
                   _Float16* hb0, _Float16* hb1, int lane) {
  v8f zero = {0.f, 0.f, 0.f, 0.f, 0.f, 0.f, 0.f, 0.f};
  v8f acc[4] = {zero, zero, zero, zero};
#pragma unroll
  for (int kb = 0; kb < NKB1; ++kb) {
    v16h a = load_a_frag(aptr[kb], lane);
#pragma unroll
    for (int nb = 0; nb < 4; ++nb)
      acc[nb] = wmma16(a, load_b_swz(W1, 4, kb, nb, lane), acc[nb]);
  }
#pragma unroll
  for (int nb = 0; nb < 4; ++nb)
    epi_store(acc[nb], b1, nb * 16, true, hb0, HB_LD, lane);
  {
    const _Float16* hrow = hb0 + (lane & 15) * HB_LD;
    v8f a2[4] = {zero, zero, zero, zero};
#pragma unroll
    for (int kb = 0; kb < 2; ++kb) {
      v16h a = load_a_frag(hrow + kb * 32, lane);
#pragma unroll
      for (int nb = 0; nb < 4; ++nb)
        a2[nb] = wmma16(a, load_b_swz(W2, 4, kb, nb, lane), a2[nb]);
    }
#pragma unroll
    for (int nb = 0; nb < 4; ++nb)
      epi_store(a2[nb], b2, nb * 16, true, hb1, HB_LD, lane);
  }
  {
    const _Float16* hrow = hb1 + (lane & 15) * HB_LD;
    v8f a3[2] = {zero, zero};
#pragma unroll
    for (int kb = 0; kb < 2; ++kb) {
      v16h a = load_a_frag(hrow + kb * 32, lane);
#pragma unroll
      for (int nb = 0; nb < 2; ++nb)
        a3[nb] = wmma16(a, load_b_swz(W3, 2, kb, nb, lane), a3[nb]);
    }
#pragma unroll
    for (int nb = 0; nb < 2; ++nb)
      epi_store(a3[nb], b3, nb * 16, true, hb0, HB_LD, lane);
  }
}

// Per-node attention: e (rows 0..30 valid in eb) -> softmax(logits/T) -> agg row.
DEVINL void attention_node(const _Float16* eb, const float* aggW, float aggb,
                           float* attnbuf, _Float16* aggrow, int lane) {
  float logit = aggb;
  const _Float16* er = eb + lane * EB_LD;
#pragma unroll
  for (int n = 0; n < 64; ++n) logit += (float)er[n] * aggW[n];
  bool valid = lane < 31;
  float l = valid ? logit : -1e30f;
  float m = l;
#pragma unroll
  for (int s = 16; s >= 1; s >>= 1) m = fmaxf(m, __shfl_xor(m, s, 32));
  float p = valid ? __expf((l - m) * INV_TEMP) : 0.0f;
  float sum = p;
#pragma unroll
  for (int s = 16; s >= 1; s >>= 1) sum += __shfl_xor(sum, s, 32);
  attnbuf[lane] = p / sum;
  float a0 = 0.0f, a1 = 0.0f;
  for (int r = 0; r < 31; ++r) {
    float w = attnbuf[r];
    a0 += w * (float)eb[r * EB_LD + lane];
    a1 += w * (float)eb[r * EB_LD + 32 + lane];
  }
  aggrow[lane]      = (_Float16)a0;
  aggrow[32 + lane] = (_Float16)a1;
}

__global__ __launch_bounds__(256)
void rld_fused_kernel(Params P, float* __restrict__ out) {
  extern __shared__ unsigned char smem[];
  const int tid  = threadIdx.x;
  const int lane = tid & 31;
  const int wave = tid >> 5;
  const int b    = blockIdx.x;

  _Float16* sSrc = (_Float16*)(smem + OFF_SRC);
  _Float16* sZ16 = (_Float16*)(smem + OFF_Z16);
  _Float16* sAgg = (_Float16*)(smem + OFF_AGG);
  _Float16* sHb  = (_Float16*)(smem + OFF_HB);
  _Float16* sEb  = (_Float16*)(smem + OFF_EB);
  float*    sZf  = (float*)(smem + OFF_ZF);
  float*    sAt  = (float*)(smem + OFF_ATTN);
  float*    sB   = (float*)(smem + OFF_BIAS);

  _Float16* sWe1 = (_Float16*)(smem + OFF_WE1);
  _Float16* sWe2 = (_Float16*)(smem + OFF_WE2);
  _Float16* sWe3 = (_Float16*)(smem + OFF_WE3);
  _Float16* sWn1 = (_Float16*)(smem + OFF_WN1);
  _Float16* sWn2 = (_Float16*)(smem + OFF_WN2);
  _Float16* sWn3 = (_Float16*)(smem + OFF_WN3);
  _Float16* sWn4 = (_Float16*)(smem + OFF_WN4);
  _Float16* sWl1 = (_Float16*)(smem + OFF_WL1);
  _Float16* sWl2 = (_Float16*)(smem + OFF_WL2);
  _Float16* sWl3 = (_Float16*)(smem + OFF_WL3);
  _Float16* sWt1 = (_Float16*)(smem + OFF_WT1);
  _Float16* sWt2 = (_Float16*)(smem + OFF_WT2);
  _Float16* sWt3 = (_Float16*)(smem + OFF_WT3);
  _Float16* sWt4 = (_Float16*)(smem + OFF_WT4);

  // ---- Stage 0: inputs + weights into LDS ---------------------------------
  const float* zb = P.p[0] + (size_t)b * (Tn * On * En);
  for (int i = tid; i < On * TEn; i += 256) {
    int o = i >> 7, c = i & 127;
    int t = c >> 4, e = c & 15;
    sSrc[o * SRC_LD + c] = (_Float16)zb[(t * On + o) * En + e];
  }
  for (int i = tid; i < On * En; i += 256) {
    int o = i >> 4, e = i & 15;
    sZf[o * EIn + e] = zb[((Tn - 1) * On + o) * En + e];  // explicit part of z0
  }
  load_w_swz(P.p[1], 256, 64, sWe1, tid);
  load_w_swz(P.p[3], 64, 64, sWe2, tid);
  load_w_swz(P.p[5], 64, 64, sWe3, tid);
  load_w_swz(P.p[9], 192, 64, sWn1, tid);
  load_w_swz(P.p[11], 64, 64, sWn2, tid);
  load_w_swz(P.p[13], 64, 32, sWn3, tid);
  load_w_swz(P.p[15], 32, 16, sWn4, tid);
  load_w_swz(P.p[17], 64, 64, sWl1, tid);
  load_w_swz(P.p[19], 64, 64, sWl2, tid);
  load_w_swz(P.p[21], 64, 64, sWl3, tid);
  load_w_swz(P.p[25], 96, 64, sWt1, tid);
  load_w_swz(P.p[27], 64, 64, sWt2, tid);
  load_w_swz(P.p[29], 64, 32, sWt3, tid);
  load_w_swz(P.p[31], 32, 32, sWt4, tid);
  copy_f32(P.p[2],  sB + BE1, 64, tid);
  copy_f32(P.p[4],  sB + BE2, 64, tid);
  copy_f32(P.p[6],  sB + BE3, 64, tid);
  copy_f32(P.p[7],  sB + AGGW, 64, tid);
  copy_f32(P.p[8],  sB + AGGB, 1, tid);
  copy_f32(P.p[10], sB + BN1, 64, tid);
  copy_f32(P.p[12], sB + BN2, 64, tid);
  copy_f32(P.p[14], sB + BN3, 32, tid);
  copy_f32(P.p[16], sB + BN4, 16, tid);
  copy_f32(P.p[18], sB + BL1, 64, tid);
  copy_f32(P.p[20], sB + BL2, 64, tid);
  copy_f32(P.p[22], sB + BL3, 64, tid);
  copy_f32(P.p[23], sB + LAGGW, 64, tid);
  copy_f32(P.p[24], sB + LAGGB, 1, tid);
  copy_f32(P.p[26], sB + BT1, 64, tid);
  copy_f32(P.p[28], sB + BT2, 64, tid);
  copy_f32(P.p[30], sB + BT3, 32, tid);
  copy_f32(P.p[32], sB + BT4, 32, tid);
  __syncthreads();

  _Float16* hb0 = sHb + wave * (2 * 16 * HB_LD);
  _Float16* hb1 = hb0 + 16 * HB_LD;
  _Float16* eb  = sEb + wave * (32 * EB_LD);
  float*    atn = sAt + wave * 32;

  // ---- Phase 1: edge MLP + attention per node (each wave owns 4 nodes) ----
  for (int i = 0; i < 4; ++i) {
    int o = wave * 4 + i;
#pragma unroll
    for (int t = 0; t < 2; ++t) {
      int r = t * 16 + (lane & 15);
      int jj = r + (r >= o ? 1 : 0);
      if (jj > 31) jj = 31;  // row 31 of tile 1 is padding, excluded by attention
      const _Float16* aptr[8];
#pragma unroll
      for (int kb = 0; kb < 8; ++kb)
        aptr[kb] = (kb < 4) ? (sSrc + o * SRC_LD + kb * 32)
                            : (sSrc + jj * SRC_LD + (kb - 4) * 32);
      pair_mlp3<8>(aptr, sWe1, sB + BE1, sWe2, sB + BE2, sWe3, sB + BE3,
                   hb0, hb1, eb + t * 16 * EB_LD, lane);
    }
    attention_node(eb, sB + AGGW, sB[AGGB], atn, sAgg + o * AGG_LD, lane);
  }
  __syncthreads();

  // ---- Phase 1b: node encoder -> z_impl (waves 0,1 each do one 16-row tile)
  if (wave < 2) {
    int t = wave;
    int o = t * 16 + (lane & 15);
    const _Float16* aptr[6];
#pragma unroll
    for (int kb = 0; kb < 6; ++kb)
      aptr[kb] = (kb < 4) ? (sSrc + o * SRC_LD + kb * 32)
                          : (sAgg + o * AGG_LD + (kb - 4) * 32);
    trunk3<6>(aptr, sWn1, sB + BN1, sWn2, sB + BN2, sWn3, sB + BN3, hb0, hb1, lane);
    // layer 4: 32 -> 16, no relu
    v8f zero = {0.f, 0.f, 0.f, 0.f, 0.f, 0.f, 0.f, 0.f};
    v8f c4 = zero;
    {
      v16h a = load_a_frag(hb0 + (lane & 15) * HB_LD, lane);
      c4 = wmma16(a, load_b_swz(sWn4, 1, 0, 0, lane), c4);
    }
    int n = lane & 15;
    int mb = (lane & 16) ? 8 : 0;
    float bv = sB[BN4 + n];
#pragma unroll
    for (int j = 0; j < 8; ++j) {
      int oo = t * 16 + mb + j;
      sZf[oo * EIn + 16 + n] = c4[j] + bv;  // implicit part of z0
    }
  }
  __syncthreads();
  for (int i = tid; i < On * EIn; i += 256) {
    int o = i >> 5, c = i & 31;
    sZ16[o * Z_LD + c] = (_Float16)sZf[i];
  }
  __syncthreads();

  // ---- Phase 2: t_future rollout ------------------------------------------
  float* outb = out + (size_t)b * (TFn * On * En);
  for (int s = 0; s < TFn; ++s) {
    // latent pair MLP + attention
    for (int i = 0; i < 4; ++i) {
      int o = wave * 4 + i;
#pragma unroll
      for (int t = 0; t < 2; ++t) {
        int r = t * 16 + (lane & 15);
        int jj = r + (r >= o ? 1 : 0);
        if (jj > 31) jj = 31;
        const _Float16* aptr[2] = {sZ16 + o * Z_LD, sZ16 + jj * Z_LD};
        pair_mlp3<2>(aptr, sWl1, sB + BL1, sWl2, sB + BL2, sWl3, sB + BL3,
                     hb0, hb1, eb + t * 16 * EB_LD, lane);
      }
      attention_node(eb, sB + LAGGW, sB[LAGGB], atn, sAgg + o * AGG_LD, lane);
    }
    __syncthreads();

    // latent transition: z += MLP([z, eagg]); emit first E columns
    if (wave < 2) {
      int t = wave;
      int o = t * 16 + (lane & 15);
      const _Float16* aptr[3] = {sZ16 + o * Z_LD, sAgg + o * AGG_LD,
                                 sAgg + o * AGG_LD + 32};
      trunk3<3>(aptr, sWt1, sB + BT1, sWt2, sB + BT2, sWt3, sB + BT3, hb0, hb1, lane);
      // layer 4: 32 -> 32, no relu
      v8f zero = {0.f, 0.f, 0.f, 0.f, 0.f, 0.f, 0.f, 0.f};
      v8f c4[2] = {zero, zero};
      {
        v16h a = load_a_frag(hb0 + (lane & 15) * HB_LD, lane);
#pragma unroll
        for (int nb = 0; nb < 2; ++nb)
          c4[nb] = wmma16(a, load_b_swz(sWt4, 2, 0, nb, lane), c4[nb]);
      }
      int n = lane & 15;
      int mb = (lane & 16) ? 8 : 0;
#pragma unroll
      for (int nb = 0; nb < 2; ++nb) {
        int col = nb * 16 + n;
        float bv = sB[BT4 + col];
#pragma unroll
        for (int j = 0; j < 8; ++j) {
          int oo = t * 16 + mb + j;
          float v = sZf[oo * EIn + col] + c4[nb][j] + bv;
          sZf[oo * EIn + col] = v;
          if (col < En) outb[((size_t)s * On + oo) * En + col] = v;
        }
      }
    }
    __syncthreads();
    for (int i = tid; i < On * EIn; i += 256) {
      int o = i >> 5, c = i & 31;
      sZ16[o * Z_LD + c] = (_Float16)sZf[i];
    }
    __syncthreads();
  }
}

extern "C" void kernel_launch(void* const* d_in, const int* in_sizes, int n_in,
                              void* d_out, int out_size, void* d_ws, size_t ws_size,
                              hipStream_t stream) {
  (void)in_sizes; (void)out_size; (void)d_ws; (void)ws_size;
  // Dynamic LDS exceeds the default cap; raise it (not a stream op, graph-safe).
  (void)hipFuncSetAttribute((const void*)rld_fused_kernel,
                            hipFuncAttributeMaxDynamicSharedMemorySize,
                            (int)SMEM_BYTES);
  Params P;
  for (int i = 0; i < 33; ++i) P.p[i] = (i < n_in) ? (const float*)d_in[i] : nullptr;
  rld_fused_kernel<<<dim3(Bn), dim3(256), SMEM_BYTES, stream>>>(P, (float*)d_out);
}
